// PrototypeSubspace_60868276519677
// MI455X (gfx1250) — matrix operand order repaired and checked by reading, not compiled, not run on hardware
//
#include <hip/hip_runtime.h>
#include <hip/hip_bf16.h>

#define B_DIM 1024
#define C_DIM 256
#define S_DIM 64
#define H_DIM 1024

typedef __attribute__((ext_vector_type(16))) __bf16 v16bf;
typedef __attribute__((ext_vector_type(8)))  __bf16 v8bf;
typedef __attribute__((ext_vector_type(8)))  float  v8f;
typedef __attribute__((ext_vector_type(4)))  float  v4f;
typedef __attribute__((ext_vector_type(2)))  float  v2f;

// Workspace layout (bytes):
//   [WS_SUB)  bf16 subspaces : C*S*H*2 = 32 MB
//   [WS_X)    bf16 x         : B*H*2   =  2 MB
//   [WS_G)    f32 Gram       : C*S*S*4 =  4 MB
//   [WS_X2)   f32 ||x||^2    : B*4     =  4 KB
static constexpr size_t WS_SUB = 0;
static constexpr size_t WS_X   = WS_SUB + (size_t)C_DIM * S_DIM * H_DIM * 2;
static constexpr size_t WS_G   = WS_X   + (size_t)B_DIM * H_DIM * 2;
static constexpr size_t WS_X2  = WS_G   + (size_t)C_DIM * S_DIM * S_DIM * 4;

// A-frag 16x32 bf16 (wave32): lane-half K split {h*8..+7} U {16+h*8..+7}
static __device__ __forceinline__ v16bf load_a_frag(const __bf16* __restrict__ row,
                                                    int kb, int half) {
    v8bf lo = *(const v8bf*)(row + kb + half * 8);
    v8bf hi = *(const v8bf*)(row + kb + 16 + half * 8);
    return __builtin_shufflevector(lo, hi, 0,1,2,3,4,5,6,7,8,9,10,11,12,13,14,15);
}
// B-frag 32x16 bf16: lane holds one column (a contiguous source row), K split {h*8..+7} U {16+h*8..+7}
static __device__ __forceinline__ v16bf load_b_frag(const __bf16* __restrict__ row,
                                                    int kb, int half) {
    v8bf lo = *(const v8bf*)(row + kb + half * 8);
    v8bf hi = *(const v8bf*)(row + kb + 16 + half * 8);
    return __builtin_shufflevector(lo, hi, 0,1,2,3,4,5,6,7,8,9,10,11,12,13,14,15);
}

#define WMMA_BF16(acc, a, b) \
    __builtin_amdgcn_wmma_f32_16x16x32_bf16(false, (a), false, (b), (short)0, (acc), false, false)

// ---------------------------------------------------------------- convert f32 -> bf16
__global__ void ps_cvt_kernel(const float* __restrict__ x, const float* __restrict__ sub,
                              __bf16* __restrict__ xb, __bf16* __restrict__ subb) {
    size_t i      = (size_t)blockIdx.x * blockDim.x + threadIdx.x;
    size_t stride = (size_t)gridDim.x * blockDim.x;
    const size_t nsub = (size_t)C_DIM * S_DIM * H_DIM;
    const size_t nx   = (size_t)B_DIM * H_DIM;
    for (size_t k = i; k < nsub; k += stride) subb[k] = (__bf16)sub[k];
    for (size_t k = i; k < nx;   k += stride) xb[k]   = (__bf16)x[k];
}

// ---------------------------------------------------------------- ||x_b||^2 (f32, one wave per row)
__global__ __launch_bounds__(32) void ps_x2_kernel(const float* __restrict__ x,
                                                   float* __restrict__ x2) {
    int b = blockIdx.x;
    int lane = threadIdx.x;
    const float* row = x + (size_t)b * H_DIM;
    float s = 0.f;
    for (int k = lane; k < H_DIM; k += 32) { float v = row[k]; s += v * v; }
    for (int off = 16; off > 0; off >>= 1) s += __shfl_xor(s, off, 32);
    if (lane == 0) x2[b] = s;
}

// Fully unrolled K=1024 bf16 WMMA strip: 16 rows x 64 cols, 4 accumulator chains.
// First K-step peeled with inline-0 C operand (no accumulator zero-init movs).
// Straight-line code: no loop-carried register constraints, all addresses fold
// into 24-bit immediate offsets, scheduler free to hoist loads deep.
static __device__ __forceinline__ void gemm_strip_k1024(const __bf16* __restrict__ arow,
                                                        const __bf16* __restrict__ brow,
                                                        int half, v8f acc[4]) {
    {
        v16bf a = load_a_frag(arow, 0, half);
        #pragma unroll
        for (int j = 0; j < 4; ++j) {
            v16bf b = load_b_frag(brow + (size_t)j * 16 * H_DIM, 0, half);
            acc[j] = WMMA_BF16((v8f){}, a, b);
        }
    }
    #pragma unroll
    for (int kb = 32; kb < H_DIM; kb += 32) {
        v16bf a = load_a_frag(arow, kb, half);
        #pragma unroll
        for (int j = 0; j < 4; ++j) {
            v16bf b = load_b_frag(brow + (size_t)j * 16 * H_DIM, kb, half);
            acc[j] = WMMA_BF16(acc[j], a, b);
        }
    }
}

// ---------------------------------------------------------------- G_c = sub_c @ sub_c^T
// One wave per 16-row strip, 4 column tiles. grid = C*4.
__global__ __launch_bounds__(32) void ps_gram_kernel(const __bf16* __restrict__ subb,
                                                     float* __restrict__ G) {
    int c    = blockIdx.x >> 2;
    int ti   = blockIdx.x & 3;
    int lane = threadIdx.x & 31;
    int half = lane >> 4, r = lane & 15;

    const __bf16* arow = subb + ((size_t)c * S_DIM + ti * 16 + r) * H_DIM;
    const __bf16* brow = subb + ((size_t)c * S_DIM + r) * H_DIM;

    v8f acc[4];
    gemm_strip_k1024(arow, brow, half, acc);

    float* gt = G + (size_t)c * S_DIM * S_DIM;
    #pragma unroll
    for (int j = 0; j < 4; ++j)
        #pragma unroll
        for (int v = 0; v < 8; ++v)
            gt[(ti * 16 + v + half * 8) * S_DIM + j * 16 + r] = acc[j][v];
}

// ---------------------------------------------------------------- fused coeff + quadratic form
// Block = (64-row b-tile, class c); wave w owns rows 16w..16w+15 across all 64 columns.
// logits[b,c] = -( coeff^T G coeff - 2*||coeff||^2 + ||x||^2 )
__global__ __launch_bounds__(128) void ps_fused_kernel(const __bf16* __restrict__ xb,
                                                       const __bf16* __restrict__ subb,
                                                       const float* __restrict__ G,
                                                       const float* __restrict__ x2,
                                                       float* __restrict__ out) {
    __shared__ float Gs[S_DIM * S_DIM];   // 16 KB: G_c (symmetric)
    __shared__ float Cs[64 * S_DIM];      // 16 KB: coeff tile [64 x 64]
    __shared__ float qs[64];
    __shared__ float ns[64];

    int bt = blockIdx.x / C_DIM;
    int c  = blockIdx.x % C_DIM;
    int b0 = bt * 64;
    int tid  = threadIdx.x;
    int wid  = tid >> 5;
    int lane = tid & 31;
    int half = lane >> 4, r = lane & 15;

    // Stage G_c into LDS (each thread 8x float4).
    {
        const v4f* gsrc = (const v4f*)(G + (size_t)c * S_DIM * S_DIM);
        v4f* gdst = (v4f*)Gs;
        #pragma unroll
        for (int i = 0; i < 8; ++i) gdst[tid + 128 * i] = gsrc[tid + 128 * i];
    }

    // ---- Stage 1: coeff strip = X[rows,:] @ sub_c^T, K = 1024.
    const __bf16* arow = xb + (size_t)(b0 + wid * 16 + r) * H_DIM;
    const __bf16* brow = subb + ((size_t)c * S_DIM + r) * H_DIM;
    v8f acc[4];
    gemm_strip_k1024(arow, brow, half, acc);

    // Spill coeff strip to LDS (stage-2 transpose source); n2 via shfl reduction.
    {
        float npart[8] = {0.f, 0.f, 0.f, 0.f, 0.f, 0.f, 0.f, 0.f};
        #pragma unroll
        for (int j = 0; j < 4; ++j)
            #pragma unroll
            for (int v = 0; v < 8; ++v) {
                float d = acc[j][v];
                Cs[(wid * 16 + v + half * 8) * S_DIM + j * 16 + r] = d;
                npart[v] += d * d;
            }
        #pragma unroll
        for (int v = 0; v < 8; ++v) {
            float s = npart[v];
            for (int off = 8; off > 0; off >>= 1) s += __shfl_xor(s, off, 32);
            if (r == 0) ns[wid * 16 + v + half * 8] = s;
        }
    }
    __syncthreads();

    // ---- Stage 2: T = Cs[rows,:] @ Gs (f32 WMMA 16x16x4, K=64).
    // G symmetric => B-frag pair {G[k][n],G[k+1][n]} = contiguous {G[n][k],G[n][k+1]}: ds_load_b64.
    v8f T[4] = {{}, {}, {}, {}};
    #pragma unroll
    for (int kb = 0; kb < S_DIM; kb += 4) {
        v2f a2 = *(const v2f*)(&Cs[(wid * 16 + r) * S_DIM + kb + half * 2]);
        #pragma unroll
        for (int j = 0; j < 4; ++j) {
            v2f b2 = *(const v2f*)(&Gs[(j * 16 + r) * S_DIM + kb + half * 2]);
            T[j] = __builtin_amdgcn_wmma_f32_16x16x4_f32(false, a2, false, b2, (short)0,
                                                         T[j], false, false);
        }
    }
    // q[m] = sum_n T[m,n]*coeff[m,n]: T and acc share the exact C/D (row,col) mapping,
    // so this is a pure in-register elementwise FMA + shfl reduction (no LDS reads).
    {
        float qpart[8] = {0.f, 0.f, 0.f, 0.f, 0.f, 0.f, 0.f, 0.f};
        #pragma unroll
        for (int j = 0; j < 4; ++j)
            #pragma unroll
            for (int v = 0; v < 8; ++v)
                qpart[v] += T[j][v] * acc[j][v];
        #pragma unroll
        for (int v = 0; v < 8; ++v) {
            float s = qpart[v];
            for (int off = 8; off > 0; off >>= 1) s += __shfl_xor(s, off, 32);
            if (r == 0) qs[wid * 16 + v + half * 8] = s;
        }
    }
    __syncthreads();

    if (tid < 64) {
        int b = b0 + tid;
        out[(size_t)b * C_DIM + c] = -(qs[tid] - 2.f * ns[tid] + x2[b]);
    }
}

// ---------------------------------------------------------------- launcher
extern "C" void kernel_launch(void* const* d_in, const int* in_sizes, int n_in,
                              void* d_out, int out_size, void* d_ws, size_t ws_size,
                              hipStream_t stream) {
    (void)in_sizes; (void)n_in; (void)out_size; (void)ws_size;
    const float* x   = (const float*)d_in[0];
    const float* sub = (const float*)d_in[1];
    char* ws = (char*)d_ws;
    __bf16* subb = (__bf16*)(ws + WS_SUB);
    __bf16* xb   = (__bf16*)(ws + WS_X);
    float*  G    = (float*)(ws + WS_G);
    float*  x2   = (float*)(ws + WS_X2);
    float*  out  = (float*)d_out;

    ps_cvt_kernel<<<1024, 256, 0, stream>>>(x, sub, xb, subb);
    ps_x2_kernel<<<B_DIM, 32, 0, stream>>>(x, x2);
    ps_gram_kernel<<<C_DIM * 4, 32, 0, stream>>>(subb, G);
    ps_fused_kernel<<<(B_DIM / 64) * C_DIM, 128, 0, stream>>>(xb, subb, G, x2, out);
}
